// PartitionalClustering_62646392979770
// MI455X (gfx1250) — matrix operand order repaired and checked by reading, not compiled, not run on hardware
//
#include <hip/hip_runtime.h>
#include <hip/hip_bf16.h>
#include <math.h>

// Problem constants (match reference)
#define BB 8
#define KK 64
#define DD 512
#define NN 4096

typedef __bf16 bf16x16 __attribute__((ext_vector_type(16)));
typedef float  f32x8   __attribute__((ext_vector_type(8)));

// ---------------------------------------------------------------------------
// WMMA fragment layouts for V_WMMA_F32_16X16X32_BF16 (wave32, ISA 7.12.2)
//
// A (16x32, MxK): lane L -> M = L%16, half h = L/16.
//   element j -> K = (j/8)*16 + h*8 + (j%8)   => two contiguous 8-elem runs
//   at byte offsets (h*8)*2 and (16+h*8)*2 within the row.
// B (32x16, KxN): lane L -> N = L%16.
//   element j -> K = (j/8)*16 + (L/16)*8 + (j%8) => same two 8-elem runs as A.
// C/D (16x16 f32): lane L -> N = L%16; vgpr i -> M = i + (L/16)*8
// ---------------------------------------------------------------------------

// bf16 source: pure 128-bit bit-copies, no conversion VALU.
__device__ inline bf16x16 load_a_frag_bf16(const __bf16* src, int ld, int row0, int k0, int lane) {
    const int m = row0 + (lane & 15);
    const int h = lane >> 4;
    const __bf16* p = src + (size_t)m * ld + k0 + (h << 3);
    union { uint4 q[2]; bf16x16 v; } u;
    u.q[0] = *(const uint4*)(p);        // K = h*8 + 0..7
    u.q[1] = *(const uint4*)(p + 16);   // K = 16 + h*8 + 0..7
    return u.v;
}

// B supplied "transposed": srcT[col, k] row-major so contraction is contiguous.
__device__ inline bf16x16 load_b_frag_bf16(const __bf16* srcT, int ld, int k0, int col0, int lane) {
    const int c = col0 + (lane & 15);
    const __bf16* p = srcT + (size_t)c * ld + k0 + ((lane >> 4) << 3);
    union { uint4 q[2]; bf16x16 v; } u;
    u.q[0] = *(const uint4*)(p);
    u.q[1] = *(const uint4*)(p + 16);
    return u.v;
}

// fp32 source variants (convert on the fly) for the small projection GEMM.
__device__ inline bf16x16 load_b_fragT_f32(const float* srcT, int ld, int k0, int col0, int lane) {
    const int c = col0 + (lane & 15);
    const float* p = srcT + (size_t)c * ld + k0 + ((lane >> 4) << 3);
    bf16x16 b;
#pragma unroll
    for (int j = 0; j < 16; ++j) b[j] = (__bf16)p[((j >> 3) << 4) + (j & 7)];
    return b;
}

__device__ inline bf16x16 load_a_frag_bias_f32(const float* X, const float* rowBias, int ld,
                                               int row0, int k0, int lane) {
    const int r = row0 + (lane & 15);
    const int k = r % KK;              // bias row (cb broadcast over batch)
    const int h = lane >> 4;
    const float* px = X + (size_t)r * ld + k0;
    const float* pb = rowBias + (size_t)k * ld + k0;
    bf16x16 a;
#pragma unroll
    for (int j = 0; j < 16; ++j) {
        const int kk = ((j >> 3) << 4) + (h << 3) + (j & 7);
        a[j] = (__bf16)(px[kk] + pb[kk]);
    }
    return a;
}

__device__ inline float silu_f(float x) { return x / (1.0f + __expf(-x)); }
__device__ inline float sigmoid_f(float x) { return 1.0f / (1.0f + __expf(-x)); }

#define WMMA_BF16(A, Bf, C) \
    __builtin_amdgcn_wmma_f32_16x16x32_bf16(false, (A), false, (Bf), (short)0, (C), false, false)

// ---------------------------------------------------------------------------
// Kernel 1: Y = silu((X + rowBias) @ W^T + colBias), rows = B*K, cols = D
// One 16x64 strip per wave: A fragment loaded once, 4 WMMAs per k-step.
// ---------------------------------------------------------------------------
__global__ void proj_silu_wmma(const float* __restrict__ X, const float* __restrict__ rowBias,
                               const float* __restrict__ W, const float* __restrict__ colBias,
                               float* __restrict__ Y) {
    const int lane = threadIdx.x & 31;
    const int wave = threadIdx.x >> 5;
    const int task = blockIdx.x * 8 + wave;             // 256 strips
    const int rt = task >> 3;                           // 32 row tiles
    const int ct0 = (task & 7) * 4;                     // 8 strips of 4 col tiles

    f32x8 acc[4] = {{}, {}, {}, {}};
#pragma unroll 2
    for (int d0 = 0; d0 < DD; d0 += 32) {
        if (d0 + 32 < DD)
            __builtin_prefetch(X + (size_t)(rt * 16 + (lane & 15)) * DD + d0 + 32, 0, 3);
        bf16x16 a = load_a_frag_bias_f32(X, rowBias, DD, rt * 16, d0, lane);
#pragma unroll
        for (int s = 0; s < 4; ++s) {
            bf16x16 b = load_b_fragT_f32(W, DD, d0, (ct0 + s) * 16, lane);
            acc[s] = WMMA_BF16(a, b, acc[s]);
        }
    }

#pragma unroll
    for (int s = 0; s < 4; ++s) {
        const int e = (ct0 + s) * 16 + (lane & 15);
        const float bce = colBias[e];
#pragma unroll
        for (int i = 0; i < 8; ++i) {
            const int r = rt * 16 + i + ((lane >> 4) << 3);
            Y[(size_t)r * DD + e] = silu_f(acc[s][i] + bce);
        }
    }
}

// ---------------------------------------------------------------------------
// Kernel 2: row-normalize c (fp32 [B*K, D]) -> cn (bf16)
// ---------------------------------------------------------------------------
__global__ void rownorm_to_bf16(const float* __restrict__ src, __bf16* __restrict__ dst) {
    __shared__ float red[128];
    const int row = blockIdx.x;
    const int tid = threadIdx.x;
    float s = 0.f;
    for (int d = tid; d < DD; d += 128) {
        const float v = src[(size_t)row * DD + d];
        s += v * v;
    }
    red[tid] = s;
    __syncthreads();
    for (int off = 64; off > 0; off >>= 1) {
        if (tid < off) red[tid] += red[tid + off];
        __syncthreads();
    }
    const float scale = 1.0f / fmaxf(sqrtf(red[0]), 1e-12f);
    for (int d = tid; d < DD; d += 128)
        dst[(size_t)row * DD + d] = (__bf16)(src[(size_t)row * DD + d] * scale);
}

// ---------------------------------------------------------------------------
// Kernel 3: per-point column norms of points [B, D, N] (over D)
// ---------------------------------------------------------------------------
__global__ void point_colnorm(const float* __restrict__ points, float* __restrict__ pnorm) {
    const int t = blockIdx.x * blockDim.x + threadIdx.x;
    if (t >= BB * NN) return;
    const int b = t / NN, n = t % NN;
    float s = 0.f;
    for (int d = 0; d < DD; ++d) {
        const float v = points[((size_t)b * DD + d) * NN + n];
        s += v * v;
    }
    pnorm[t] = fmaxf(sqrtf(s), 1e-12f);
}

// ---------------------------------------------------------------------------
// Kernel 4: batched tile transpose fp32 [R,C] -> bf16 [C,R], optional per-col scale
// ---------------------------------------------------------------------------
__global__ void transpose_conv_bf16(const float* __restrict__ src, __bf16* __restrict__ dst,
                                    int R, int C, const float* __restrict__ colNorm) {
    __shared__ float tile[32][33];
    const int c0 = blockIdx.x * 32;
    const int r0 = blockIdx.y * 32;
    const int b = blockIdx.z;
    const float* s = src + (size_t)b * R * C;
    for (int rr = threadIdx.y; rr < 32; rr += 8)
        tile[rr][threadIdx.x] = s[(size_t)(r0 + rr) * C + c0 + threadIdx.x];
    __syncthreads();
    __bf16* d = dst + (size_t)b * R * C;
    for (int cc = threadIdx.y; cc < 32; cc += 8) {
        const int c = c0 + cc;
        float sc = 1.0f;
        if (colNorm) sc = 1.0f / fmaxf(colNorm[(size_t)b * C + c], 1e-12f);
        d[(size_t)c * R + r0 + threadIdx.x] = (__bf16)(tile[threadIdx.x][cc] * sc);
    }
}

// ---------------------------------------------------------------------------
// Kernel 5: sim_ = alpha[k] * (cn @ pn) + beta[k]   -> [B, K, N] fp32
// One 16x64 strip per wave (A reused across 4 N-tiles).
// ---------------------------------------------------------------------------
__global__ void sim_wmma(const __bf16* __restrict__ cn, const __bf16* __restrict__ pnT,
                         const float* __restrict__ alpha, const float* __restrict__ beta,
                         float* __restrict__ simOut) {
    const int lane = threadIdx.x & 31;
    const int wave = threadIdx.x >> 5;
    const int task = blockIdx.x * 8 + wave;             // 2048 strips
    const int perB = (KK / 16) * (NN / 64);             // 4 * 64 = 256
    const int b = task / perB;
    const int tt = task % perB;
    const int ktile = tt / (NN / 64);
    const int nt0 = (tt % (NN / 64)) * 4;

    const __bf16* cnB = cn + (size_t)b * KK * DD;
    const __bf16* pnB = pnT + (size_t)b * NN * DD;

    f32x8 acc[4] = {{}, {}, {}, {}};
#pragma unroll 2
    for (int d0 = 0; d0 < DD; d0 += 32) {
        if (d0 + 32 < DD) {
            __builtin_prefetch(pnB + (size_t)(nt0 * 16 + (lane & 15)) * DD + d0 + 32, 0, 3);
            __builtin_prefetch(pnB + (size_t)((nt0 + 2) * 16 + (lane & 15)) * DD + d0 + 32, 0, 3);
        }
        bf16x16 a = load_a_frag_bf16(cnB, DD, ktile * 16, d0, lane);
#pragma unroll
        for (int s = 0; s < 4; ++s) {
            bf16x16 bf = load_b_frag_bf16(pnB, DD, d0, (nt0 + s) * 16, lane);
            acc[s] = WMMA_BF16(a, bf, acc[s]);
        }
    }

#pragma unroll
    for (int s = 0; s < 4; ++s) {
        const int n = (nt0 + s) * 16 + (lane & 15);
#pragma unroll
        for (int i = 0; i < 8; ++i) {
            const int m = ktile * 16 + i + ((lane >> 4) << 3);
            simOut[((size_t)(b * KK + m)) * NN + n] = alpha[m] * acc[s][i] + beta[m];
        }
    }
}

// ---------------------------------------------------------------------------
// Kernel 6a: zero the per-cluster counts
// ---------------------------------------------------------------------------
__global__ void zero_counts(int* __restrict__ counts) {
    const int t = blockIdx.x * blockDim.x + threadIdx.x;
    if (t < BB * KK) counts[t] = 0;
}

// Kernel 6b: argmax over K, sigmoid gate, counts, dense one-hot assignment (bf16)
__global__ void argmax_assign(const float* __restrict__ sim, int* __restrict__ idx,
                              float* __restrict__ aval, int* __restrict__ counts,
                              __bf16* __restrict__ assign) {
    const int t = blockIdx.x * blockDim.x + threadIdx.x;
    if (t >= BB * NN) return;
    const int b = t / NN, n = t % NN;
    float best = -3.4e38f;
    int bi = 0;
    for (int k = 0; k < KK; ++k) {
        const float v = sim[((size_t)(b * KK + k)) * NN + n];
        if (v > best) { best = v; bi = k; }
    }
    const float a = sigmoid_f(best);
    idx[t] = bi;
    aval[t] = a;
    atomicAdd(&counts[b * KK + bi], 1);
    for (int k = 0; k < KK; ++k)
        assign[((size_t)(b * KK + k)) * NN + n] = (__bf16)((k == bi) ? a : 0.0f);
}

// ---------------------------------------------------------------------------
// Kernel 7: agg = (vc + assign @ values) / (count + 1)   -> [B, K, D] fp32
// One 16x64 strip per wave (A = assignment rows reused across 4 D-tiles).
// ---------------------------------------------------------------------------
__global__ void agg_wmma(const __bf16* __restrict__ assign, const __bf16* __restrict__ valT,
                         const float* __restrict__ vc, const int* __restrict__ counts,
                         float* __restrict__ agg) {
    const int lane = threadIdx.x & 31;
    const int wave = threadIdx.x >> 5;
    const int task = blockIdx.x * 8 + wave;             // 256 strips
    const int perB = (KK / 16) * (DD / 64);             // 4 * 8 = 32
    const int b = task / perB;
    const int tt = task % perB;
    const int ktile = tt / (DD / 64);
    const int dt0 = (tt % (DD / 64)) * 4;

    const __bf16* asB = assign + (size_t)b * KK * NN;
    const __bf16* vtB = valT + (size_t)b * DD * NN;

    f32x8 acc[4] = {{}, {}, {}, {}};
#pragma unroll 2
    for (int n0 = 0; n0 < NN; n0 += 32) {
        if (n0 + 32 < NN) {
            __builtin_prefetch(vtB + (size_t)(dt0 * 16 + (lane & 15)) * NN + n0 + 32, 0, 3);
            __builtin_prefetch(vtB + (size_t)((dt0 + 2) * 16 + (lane & 15)) * NN + n0 + 32, 0, 3);
        }
        bf16x16 a = load_a_frag_bf16(asB, NN, ktile * 16, n0, lane);
#pragma unroll
        for (int s = 0; s < 4; ++s) {
            bf16x16 bf = load_b_frag_bf16(vtB, NN, n0, (dt0 + s) * 16, lane);
            acc[s] = WMMA_BF16(a, bf, acc[s]);
        }
    }

#pragma unroll
    for (int s = 0; s < 4; ++s) {
        const int d = (dt0 + s) * 16 + (lane & 15);
#pragma unroll
        for (int i = 0; i < 8; ++i) {
            const int k = ktile * 16 + i + ((lane >> 4) << 3);
            const size_t o = ((size_t)(b * KK + k)) * DD + d;
            const float denom = (float)counts[b * KK + k] + 1.0f;
            agg[o] = (vc[o] + acc[s][i]) / denom;
        }
    }
}

// ---------------------------------------------------------------------------
// Kernel 8: x[b,d,n] = agg[b, idx[b,n], d] * aval[b,n]   (gather, coalesced in n)
// ---------------------------------------------------------------------------
__global__ void combine_gather(const float* __restrict__ agg, const int* __restrict__ idx,
                               const float* __restrict__ aval, float* __restrict__ x) {
    const size_t t = (size_t)blockIdx.x * blockDim.x + threadIdx.x;
    if (t >= (size_t)BB * DD * NN) return;
    const int n = (int)(t % NN);
    const int d = (int)((t / NN) % DD);
    const int b = (int)(t / ((size_t)DD * NN));
    const int k = idx[b * NN + n];
    x[t] = agg[((size_t)(b * KK + k)) * DD + d] * aval[b * NN + n];
}

// ---------------------------------------------------------------------------
// Host-side launch
// ---------------------------------------------------------------------------
extern "C" void kernel_launch(void* const* d_in, const int* in_sizes, int n_in,
                              void* d_out, int out_size, void* d_ws, size_t ws_size,
                              hipStream_t stream) {
    const float* points   = (const float*)d_in[0];   // [B, D, N]
    const float* clusters = (const float*)d_in[1];   // [B, K, D]
    const float* values   = (const float*)d_in[2];   // [B, N, D]
    const float* vclus    = (const float*)d_in[3];   // [B, K, D]
    const float* Wc       = (const float*)d_in[4];   // [D, D]
    const float* bc       = (const float*)d_in[5];   // [D]
    const float* cb       = (const float*)d_in[6];   // [K, D]
    const float* Wv       = (const float*)d_in[7];   // [D, D]
    const float* bv       = (const float*)d_in[8];   // [D]
    const float* vcb      = (const float*)d_in[9];   // [K, D]
    const float* salpha   = (const float*)d_in[10];  // [K]
    const float* sbeta    = (const float*)d_in[11];  // [K]
    float* x_out = (float*)d_out;                    // [B, D, N]

    // Workspace carve-out (256B aligned)
    char* ws = (char*)d_ws;
    size_t off = 0;
    auto carve = [&](size_t bytes) -> char* {
        char* p = ws + off;
        off = (off + bytes + 255) & ~(size_t)255;
        return p;
    };
    float*  c_f    = (float*) carve((size_t)BB * KK * DD * 4);   // 1 MB (reused as agg)
    float*  vc_f   = (float*) carve((size_t)BB * KK * DD * 4);   // 1 MB
    __bf16* cn_bf  = (__bf16*)carve((size_t)BB * KK * DD * 2);   // 0.5 MB
    float*  pnorm  = (float*) carve((size_t)BB * NN * 4);        // 128 KB
    int*    idx    = (int*)   carve((size_t)BB * NN * 4);        // 128 KB
    float*  aval   = (float*) carve((size_t)BB * NN * 4);        // 128 KB
    int*    counts = (int*)   carve((size_t)BB * KK * 4);        // 2 KB
    __bf16* assign = (__bf16*)carve((size_t)BB * KK * NN * 2);   // 4 MB
    __bf16* pnT    = (__bf16*)carve((size_t)BB * NN * DD * 2);   // 32 MB
    __bf16* valT   = (__bf16*)carve((size_t)BB * DD * NN * 2);   // 32 MB
    float*  sim    = (float*) carve((size_t)BB * KK * NN * 4);   // 8 MB
    (void)ws_size; (void)n_in; (void)in_sizes; (void)out_size;

    // 1) projections + SiLU (WMMA bf16, 16x64 strips)
    proj_silu_wmma<<<32, 256, 0, stream>>>(clusters, cb, Wc, bc, c_f);
    proj_silu_wmma<<<32, 256, 0, stream>>>(vclus, vcb, Wv, bv, vc_f);

    // 2) normalize c rows -> bf16
    rownorm_to_bf16<<<BB * KK, 128, 0, stream>>>(c_f, cn_bf);

    // 3) point column norms
    point_colnorm<<<(BB * NN + 255) / 256, 256, 0, stream>>>(points, pnorm);

    // 4) transposes to bf16: points [D,N] -> pnT [N,D] (scaled); values [N,D] -> valT [D,N]
    {
        dim3 g1(NN / 32, DD / 32, BB), blk(32, 8);
        transpose_conv_bf16<<<g1, blk, 0, stream>>>(points, pnT, DD, NN, pnorm);
        dim3 g2(DD / 32, NN / 32, BB);
        transpose_conv_bf16<<<g2, blk, 0, stream>>>(values, valT, NN, DD, nullptr);
    }

    // 5) sim = alpha * (cn @ pn) + beta  (WMMA bf16, 16x64 strips)
    sim_wmma<<<256, 256, 0, stream>>>(cn_bf, pnT, salpha, sbeta, sim);

    // 6) argmax assignment + counts + dense one-hot bf16 A-matrix
    zero_counts<<<2, 256, 0, stream>>>(counts);
    argmax_assign<<<(BB * NN + 255) / 256, 256, 0, stream>>>(sim, idx, aval, counts, assign);

    // 7) agg = (vc + assign @ values) / (count + 1)  (WMMA bf16, 16x64 strips)
    agg_wmma<<<32, 256, 0, stream>>>(assign, valT, vc_f, counts, c_f /*reuse as agg*/);

    // 8) combine: x[b,d,n] = agg[b, idx, d] * aval
    {
        const size_t total = (size_t)BB * DD * NN;
        combine_gather<<<(unsigned)((total + 255) / 256), 256, 0, stream>>>(c_f, idx, aval, x_out);
    }
}
